// Seq2Seq_53850299958101
// MI455X (gfx1250) — matrix operand order repaired and checked by reading, not compiled, not run on hardware
//
#include <hip/hip_runtime.h>
#include <hip/hip_bf16.h>
#include <math.h>

typedef __bf16 bf16;
typedef __attribute__((ext_vector_type(16))) __bf16 v16bf;
typedef __attribute__((ext_vector_type(8)))  __bf16 bf16x8;
typedef __attribute__((ext_vector_type(8)))  float  v8f;

// ---------------------------------------------------------------------------
// WMMA helpers (CDNA5 gfx1250, wave32). D = A(16x32 bf16) x B(32x16 bf16) + C(f32)
// A layout: row M = lane&15; lane-half g=lane>>4; elem j: K = (j>=8?16:0)+8g+(j&7)
//   -> per lane: two contiguous 8-element runs at k0+8g and k0+16+8g  (16B each)
// B layout: col N = lane&15; elem j: K = (j>=8?16:0)+8g+(j&7) (same as A)
// C/D layout: N = lane&15; elem r: M = r + 8*(lane>>4)
// All K dims padded to multiples of 32; all rows 16B aligned -> pure b128 loads.
// ---------------------------------------------------------------------------
__device__ inline v8f wmma_bf16(v16bf a, v16bf b, v8f c) {
  return __builtin_amdgcn_wmma_f32_16x16x32_bf16(false, a, false, b, (short)0, c,
                                                 false, false);
}

__device__ inline v16bf combine16(bf16x8 lo, bf16x8 hi) {
  return __builtin_shufflevector(lo, hi, 0, 1, 2, 3, 4, 5, 6, 7, 8, 9, 10, 11, 12,
                                 13, 14, 15);
}

__device__ inline v16bf load_a_bf16(const bf16* base, int ld, int row0, int k0, int lane) {
  const int r = row0 + (lane & 15);
  const int g = (lane >> 4) * 8;
  const bf16* p = base + (size_t)r * ld + k0 + g;
  bf16x8 lo = *(const bf16x8*)p;
  bf16x8 hi = *(const bf16x8*)(p + 16);
  return combine16(lo, hi);
}

__device__ inline v16bf load_b_bf16(const bf16* W, int ld, int n0, int k0, int lane) {
  const int n = n0 + (lane & 15);
  const int g = (lane >> 4) * 8;
  const bf16* p = W + (size_t)n * ld + k0 + g;
  bf16x8 lo = *(const bf16x8*)p;
  bf16x8 hi = *(const bf16x8*)(p + 16);
  return combine16(lo, hi);
}

__device__ inline float sigm(float x) { return 1.f / (1.f + __expf(-x)); }

// ---------------------------------------------------------------------------
// Weight / activation prep
// ---------------------------------------------------------------------------
__global__ void k_cvt_bf16(const float* __restrict__ s, bf16* __restrict__ d, int n) {
  int i = blockIdx.x * blockDim.x + threadIdx.x;
  if (i < n) d[i] = (bf16)s[i];
}

// row-wise convert with K padding: dst[r][k] = k<Ks ? src[r][k] : 0   (dst ld = Kd)
__global__ void k_cvt_pad(const float* __restrict__ s, bf16* __restrict__ d, int rows,
                          int Ks, int Kd) {
  int i = blockIdx.x * blockDim.x + threadIdx.x;
  if (i >= rows * Kd) return;
  int r = i / Kd, k = i - r * Kd;
  d[i] = (k < Ks) ? (bf16)s[(size_t)r * Ks + k] : (bf16)0.f;
}

// combined decoder weight: Wd[n][k], n<2048, k<2048; k<1536 -> Wih[n][k], else Whh[n][k-1536]
__global__ void k_build_dec_w(const float* __restrict__ Wih, const float* __restrict__ Whh,
                              bf16* __restrict__ Wd) {
  int i = blockIdx.x * blockDim.x + threadIdx.x;
  if (i >= 2048 * 2048) return;
  int n = i >> 11, k = i & 2047;
  float v = (k < 1536) ? Wih[(size_t)n * 1536 + k] : Whh[(size_t)n * 512 + (k - 1536)];
  Wd[i] = (bf16)v;
}

// ---------------------------------------------------------------------------
// xg GEMM: XG[m][n] = sum_k X[m][k]*W[n][k] + bih[n]+bhh[n]   (N fixed = 1024)
// X bf16 (ld = K, padded). block = 128 threads (4 waves);
// wave -> n-tile blockIdx.x*4+wv, m-tile blockIdx.y
// ---------------------------------------------------------------------------
__global__ void k_xg_gemm(const bf16* __restrict__ X, const bf16* __restrict__ W,
                          const float* __restrict__ bih, const float* __restrict__ bhh,
                          bf16* __restrict__ XG, int K) {
  const int lane = threadIdx.x & 31, wv = threadIdx.x >> 5;
  const int nt = blockIdx.x * 4 + wv;
  const int m0 = blockIdx.y * 16;
  v8f acc = {};
  for (int k0 = 0; k0 < K; k0 += 32) {
    v16bf a = load_a_bf16(X, K, m0, k0, lane);
    v16bf b = load_b_bf16(W, K, nt * 16, k0, lane);
    acc = wmma_bf16(a, b, acc);
  }
  int n = nt * 16 + (lane & 15);
  float bias = bih[n] + bhh[n];
  #pragma unroll
  for (int r = 0; r < 8; ++r) {
    int m = m0 + (lane >> 4) * 8 + r;
    XG[(size_t)m * 1024 + n] = (bf16)(acc[r] + bias);
  }
}

// ---------------------------------------------------------------------------
// comp_listener GEMM: Y[m][n] = tanh(sum_k X[m][k]*W[n][k] + bias[n]), f32 out
// ---------------------------------------------------------------------------
__global__ void k_psi_gemm(const bf16* __restrict__ X, const bf16* __restrict__ W,
                           const float* __restrict__ bias, float* __restrict__ Y,
                           int N, int K) {
  const int lane = threadIdx.x & 31, wv = threadIdx.x >> 5;
  const int nt = blockIdx.x * 4 + wv;
  if (nt * 16 >= N) return;
  const int m0 = blockIdx.y * 16;
  v8f acc = {};
  for (int k0 = 0; k0 < K; k0 += 32) {
    v16bf a = load_a_bf16(X, K, m0, k0, lane);
    v16bf b = load_b_bf16(W, K, nt * 16, k0, lane);
    acc = wmma_bf16(a, b, acc);
  }
  int n = nt * 16 + (lane & 15);
  float bs = bias[n];
  #pragma unroll
  for (int r = 0; r < 8; ++r) {
    int m = m0 + (lane >> 4) * 8 + r;
    Y[(size_t)m * N + n] = tanhf(acc[r] + bs);
  }
}

// ---------------------------------------------------------------------------
// Persistent BLSTM recurrence. grid=2 (dir), block=1024 (32 waves).
// h (32x256 bf16) in LDS; c in registers. Per step: g = xg[t] + h@Whh^T
// wave w: batch-tile mt=w&1, j-tile jt=w>>1; 4 gate accumulators at n=gate*256+j.
// Output Y[(b*T+t)*512 + dir*256 + j]  (== pBLSTM concat/reshape layout).
// ---------------------------------------------------------------------------
__global__ void __launch_bounds__(1024, 1)
k_lstm_rec(const bf16* __restrict__ XGf, const bf16* __restrict__ XGb,
           const bf16* __restrict__ Whhf, const bf16* __restrict__ Whhb,
           float* __restrict__ Y, int T) {
  __shared__ alignas(16) bf16 h_s[32 * 256];
  const int dir = blockIdx.x;
  const bf16* XG  = dir ? XGb : XGf;
  const bf16* Whh = dir ? Whhb : Whhf;
  const int tid = threadIdx.x, lane = tid & 31, wv = tid >> 5;
  const int mt = wv & 1, jt = wv >> 1;
  const int jcol = jt * 16 + (lane & 15);
  float c_r[8];
  #pragma unroll
  for (int r = 0; r < 8; ++r) c_r[r] = 0.f;
  for (int i = tid; i < 32 * 256; i += 1024) h_s[i] = (bf16)0.f;
  __syncthreads();

  for (int t = 0; t < T; ++t) {
    const int ts = dir ? (T - 1 - t) : t;
    v8f acc[4] = {};
    for (int k0 = 0; k0 < 256; k0 += 32) {
      v16bf a = load_a_bf16(h_s, 256, mt * 16, k0, lane);
      #pragma unroll
      for (int gt = 0; gt < 4; ++gt) {
        v16bf b = load_b_bf16(Whh, 256, gt * 256 + jt * 16, k0, lane);
        acc[gt] = wmma_bf16(a, b, acc[gt]);
      }
    }
    float hv[8];
    #pragma unroll
    for (int r = 0; r < 8; ++r) {
      int bb = mt * 16 + (lane >> 4) * 8 + r;
      const bf16* xg = XG + (size_t)(bb * T + ts) * 1024;
      float gi = acc[0][r] + (float)xg[jcol];
      float gf = acc[1][r] + (float)xg[256 + jcol];
      float gg = acc[2][r] + (float)xg[512 + jcol];
      float go = acc[3][r] + (float)xg[768 + jcol];
      float c  = sigm(gf) * c_r[r] + sigm(gi) * tanhf(gg);
      c_r[r] = c;
      hv[r]  = sigm(go) * tanhf(c);
      Y[(size_t)(bb * T + ts) * 512 + dir * 256 + jcol] = hv[r];
    }
    __syncthreads();
    #pragma unroll
    for (int r = 0; r < 8; ++r) {
      int bb = mt * 16 + (lane >> 4) * 8 + r;
      h_s[bb * 256 + jcol] = (bf16)hv[r];
    }
    __syncthreads();
  }
}

// ---------------------------------------------------------------------------
// Persistent attention decoder. grid=1, block=1024 (32 waves), dynamic LDS.
// ---------------------------------------------------------------------------
#define SATT 112
#define SM_XIN   0                      // bf16 [32][2048]            131072 B
#define SM_CDEC  131072                 // f32  [32][128]              16384 B
#define SM_ATT   147456                 // f32  [32][SATT]             14336 B
#define SM_CUR   161792                 // f32  [32][64]                8192 B
#define SM_AMAX  169984                 // int  [32]                     128 B
#define SMEM_DEC 170112

__global__ void __launch_bounds__(1024, 1)
k_decoder(const float* __restrict__ enc, const float* __restrict__ CL,
          const bf16* __restrict__ Wdec, const float* __restrict__ bih,
          const float* __restrict__ bhh, const bf16* __restrict__ phiW,
          const bf16* __restrict__ charW, const float* __restrict__ charB,
          const float* __restrict__ embed, const int* __restrict__ state_len,
          const int* __restrict__ decode_step_p, float* __restrict__ out) {
  extern __shared__ char smem[];
  bf16*  xin  = (bf16*)(smem + SM_XIN);    // [b][0:512]=last, [512:1536]=ctx, [1536:2048]=h
  float* cdec = (float*)(smem + SM_CDEC);
  float* att  = (float*)(smem + SM_ATT);
  float* cur  = (float*)(smem + SM_CUR);
  int*   amax = (int*)(smem + SM_AMAX);

  const int tid = threadIdx.x, lane = tid & 31, wv = tid >> 5;
  const int nstep = *decode_step_p;
  const int mt = wv & 1, jgrp = wv >> 1;

  // init: last=embed[0], ctx=0, h=0; write enc_len output
  for (int i = tid; i < 32 * 2048; i += 1024) {
    int d = i & 2047;
    xin[i] = (d < 512) ? (bf16)embed[d] : (bf16)0.f;
  }
  if (tid < 32) out[tid] = (float)(state_len[tid] / 16);
  float c_r[2][8];
  #pragma unroll
  for (int q = 0; q < 2; ++q)
    #pragma unroll
    for (int r = 0; r < 8; ++r) c_r[q][r] = 0.f;
  __syncthreads();

  for (int step = 0; step < nstep; ++step) {
    // ---- P0: comp_dec = tanh(h @ phiW^T)  (M=32,N=128,K=512) on waves 0..15
    if (wv < 16) {
      int pmt = wv & 1, pnt = wv >> 1;
      v8f acc = {};
      for (int k0 = 0; k0 < 512; k0 += 32) {
        v16bf a = load_a_bf16(xin + 1536, 2048, pmt * 16, k0, lane);
        v16bf b = load_b_bf16(phiW, 512, pnt * 16, k0, lane);
        acc = wmma_bf16(a, b, acc);
      }
      int n = pnt * 16 + (lane & 15);
      #pragma unroll
      for (int r = 0; r < 8; ++r) {
        int bb = pmt * 16 + (lane >> 4) * 8 + r;
        cdec[bb * 128 + n] = tanhf(acc[r]);
      }
    }
    __syncthreads();

    // ---- P1: energy + softmax, one wave per batch row
    {
      int b = wv;
      int elen = state_len[b] / 16;
      float ev[4];
      #pragma unroll
      for (int u = 0; u < 4; ++u) {
        int s = lane + 32 * u;
        float e = -INFINITY;
        if (s < 100 && s < elen) {
          const float* cl = CL + (size_t)(b * 100 + s) * 128;
          float a = 0.f;
          for (int k = 0; k < 128; ++k) a += cl[k] * cdec[b * 128 + k];
          e = a;
        }
        ev[u] = e;
      }
      float m = fmaxf(fmaxf(ev[0], ev[1]), fmaxf(ev[2], ev[3]));
      for (int o = 16; o > 0; o >>= 1) m = fmaxf(m, __shfl_xor(m, o, 32));
      float ssum = 0.f;
      #pragma unroll
      for (int u = 0; u < 4; ++u) {
        ev[u] = (ev[u] == -INFINITY) ? 0.f : __expf(ev[u] - m);
        ssum += ev[u];
      }
      for (int o = 16; o > 0; o >>= 1) ssum += __shfl_xor(ssum, o, 32);
      float inv = 1.f / ssum;
      #pragma unroll
      for (int u = 0; u < 4; ++u) {
        int s = lane + 32 * u;
        if (s < 100) {
          float a = ev[u] * inv;
          att[b * SATT + s] = a;
          out[32 + 32 * nstep * 64 + (size_t)(b * nstep + step) * 100 + s] = a;
        }
      }
    }
    __syncthreads();

    // ---- P2: context[b][d] = sum_s att*enc -> xin[b][512+d] (bf16)
    {
      int b = tid >> 5, d0 = tid & 31;
      #pragma unroll 1
      for (int i = 0; i < 32; ++i) {
        int d = d0 + 32 * i;
        float a = 0.f;
        for (int s = 0; s < 100; ++s)
          a += att[b * SATT + s] * enc[(size_t)(b * 100 + s) * 1024 + d];
        xin[b * 2048 + 512 + d] = (bf16)a;
      }
    }
    __syncthreads();

    // ---- P3: gates = [last,ctx,h] @ Wdec^T  (M=32, N=2048, K=2048)
    v8f acc[2][4] = {};
    for (int k0 = 0; k0 < 2048; k0 += 32) {
      v16bf a = load_a_bf16(xin, 2048, mt * 16, k0, lane);
      #pragma unroll
      for (int q = 0; q < 2; ++q)
        #pragma unroll
        for (int gt = 0; gt < 4; ++gt) {
          int n0 = gt * 512 + (jgrp * 2 + q) * 16;
          v16bf b = load_b_bf16(Wdec, 2048, n0, k0, lane);
          acc[q][gt] = wmma_bf16(a, b, acc[q][gt]);
        }
    }
    float hv[2][8];
    #pragma unroll
    for (int q = 0; q < 2; ++q) {
      int j = (jgrp * 2 + q) * 16 + (lane & 15);
      float bi_ = bih[j] + bhh[j];
      float bf_ = bih[512 + j] + bhh[512 + j];
      float bg_ = bih[1024 + j] + bhh[1024 + j];
      float bo_ = bih[1536 + j] + bhh[1536 + j];
      #pragma unroll
      for (int r = 0; r < 8; ++r) {
        float gi = acc[q][0][r] + bi_;
        float gf = acc[q][1][r] + bf_;
        float gg = acc[q][2][r] + bg_;
        float go = acc[q][3][r] + bo_;
        float c = sigm(gf) * c_r[q][r] + sigm(gi) * tanhf(gg);
        c_r[q][r] = c;
        hv[q][r] = sigm(go) * tanhf(c);
      }
    }
    __syncthreads();  // all xin reads complete before overwriting h region
    #pragma unroll
    for (int q = 0; q < 2; ++q) {
      int j = (jgrp * 2 + q) * 16 + (lane & 15);
      #pragma unroll
      for (int r = 0; r < 8; ++r) {
        int bb = mt * 16 + (lane >> 4) * 8 + r;
        xin[bb * 2048 + 1536 + j] = (bf16)hv[q][r];
      }
    }
    __syncthreads();

    // ---- P4: char logits, argmax, next "last"
    #pragma unroll
    for (int ii = 0; ii < 2; ++ii) {
      int idx = tid + 1024 * ii;
      int b = idx >> 6, v = idx & 63;
      float a = charB[v];
      for (int k = 0; k < 512; ++k)
        a += (float)xin[b * 2048 + 1536 + k] * (float)charW[v * 512 + k];
      cur[b * 64 + v] = a;
      out[32 + (size_t)(b * nstep + step) * 64 + v] = a;
    }
    __syncthreads();
    if (tid < 32) {
      int b = tid;
      float best = cur[b * 64];
      int bi = 0;
      for (int v = 1; v < 64; ++v) {
        float x = cur[b * 64 + v];
        if (x > best) { best = x; bi = v; }
      }
      amax[b] = bi;
    }
    __syncthreads();
    for (int i = tid; i < 32 * 512; i += 1024) {
      int b = i >> 9, d = i & 511;
      xin[b * 2048 + d] = (bf16)embed[(size_t)amax[b] * 512 + d];
    }
    __syncthreads();
  }
}

// ---------------------------------------------------------------------------
// Host: carve workspace, convert weights, run 4 pBLSTM layers + decoder.
// ---------------------------------------------------------------------------
extern "C" void kernel_launch(void* const* d_in, const int* in_sizes, int n_in,
                              void* d_out, int out_size, void* d_ws, size_t ws_size,
                              hipStream_t stream) {
  (void)in_sizes; (void)n_in; (void)out_size; (void)ws_size;
  const float* audio       = (const float*)d_in[0];
  const int*   decode_step = (const int*)d_in[1];
  const int*   state_len   = (const int*)d_in[2];
  int p = 3;
  const float *eWih[4][2], *eWhh[4][2], *ebih[4][2], *ebhh[4][2];
  for (int l = 0; l < 4; ++l)
    for (int d = 0; d < 2; ++d) {
      eWih[l][d] = (const float*)d_in[p + 0];
      eWhh[l][d] = (const float*)d_in[p + 1];
      ebih[l][d] = (const float*)d_in[p + 2];
      ebhh[l][d] = (const float*)d_in[p + 3];
      p += 4;
    }
  const float* dWih = (const float*)d_in[p + 0];
  const float* dWhh = (const float*)d_in[p + 1];
  const float* dbih = (const float*)d_in[p + 2];
  const float* dbhh = (const float*)d_in[p + 3];
  p += 4;
  const float* phiW  = (const float*)d_in[p++];
  const float* psiW  = (const float*)d_in[p++];
  const float* psib  = (const float*)d_in[p++];
  const float* embed = (const float*)d_in[p++];
  const float* charW = (const float*)d_in[p++];
  const float* charb = (const float*)d_in[p++];

  char* base = (char*)d_ws;
  size_t off = 0;
  auto carve = [&](size_t bytes) -> char* {
    char* r = base + off;
    off = (off + bytes + 255) & ~(size_t)255;
    return r;
  };
  bf16* WihB[4][2];
  bf16* WhhB[4][2];
  for (int l = 0; l < 4; ++l)
    for (int d = 0; d < 2; ++d) {
      WihB[l][d] = (bf16*)carve((size_t)(l == 0 ? 1024 * 96 : 1024 * 1024) * 2);
      WhhB[l][d] = (bf16*)carve((size_t)1024 * 256 * 2);
    }
  bf16*  WdecB = (bf16*)carve((size_t)2048 * 2048 * 2);
  bf16*  phiB  = (bf16*)carve((size_t)128 * 512 * 2);
  bf16*  psiB  = (bf16*)carve((size_t)128 * 1024 * 2);
  bf16*  charB = (bf16*)carve((size_t)64 * 512 * 2);
  float* bufA  = (float*)carve((size_t)32 * 1600 * 512 * 4);   // f32 layer outputs
  float* bufB  = (float*)carve((size_t)32 * 800 * 512 * 4);
  bf16*  XB    = (bf16*)carve((size_t)32 * 1600 * 512 * 2);    // bf16 GEMM A operand
  bf16*  XG0   = (bf16*)carve((size_t)32 * 1600 * 1024 * 2);
  bf16*  XG1   = (bf16*)carve((size_t)32 * 1600 * 1024 * 2);
  float* CL    = (float*)carve((size_t)3200 * 128 * 4);

  auto cv = [&](const float* s, bf16* d, int n) {
    k_cvt_bf16<<<dim3((n + 255) / 256), dim3(256), 0, stream>>>(s, d, n);
  };
  // weights: layer0 Wih padded 80 -> 96 so GEMM K loop has no tail
  for (int l = 0; l < 4; ++l)
    for (int d = 0; d < 2; ++d) {
      if (l == 0)
        k_cvt_pad<<<dim3((1024 * 96 + 255) / 256), dim3(256), 0, stream>>>(
            eWih[l][d], WihB[l][d], 1024, 80, 96);
      else
        cv(eWih[l][d], WihB[l][d], 1024 * 1024);
      cv(eWhh[l][d], WhhB[l][d], 1024 * 256);
    }
  k_build_dec_w<<<dim3((2048 * 2048 + 255) / 256), dim3(256), 0, stream>>>(dWih, dWhh, WdecB);
  cv(phiW, phiB, 128 * 512);
  cv(psiW, psiB, 128 * 1024);
  cv(charW, charB, 64 * 512);

  // audio -> bf16, padded K 80 -> 96
  k_cvt_pad<<<dim3((32 * 1600 * 96 + 255) / 256), dim3(256), 0, stream>>>(
      audio, XB, 32 * 1600, 80, 96);

  // ---- encoder: 4 pyramidal BLSTM layers
  int T = 1600, K = 96;
  float* outs[4] = {bufA, bufB, bufA, bufB};
  for (int l = 0; l < 4; ++l) {
    int M = 32 * T;
    k_xg_gemm<<<dim3(16, M / 16), dim3(128), 0, stream>>>(XB, WihB[l][0], ebih[l][0],
                                                          ebhh[l][0], XG0, K);
    k_xg_gemm<<<dim3(16, M / 16), dim3(128), 0, stream>>>(XB, WihB[l][1], ebih[l][1],
                                                          ebhh[l][1], XG1, K);
    k_lstm_rec<<<dim3(2), dim3(1024), 0, stream>>>(XG0, XG1, WhhB[l][0], WhhB[l][1],
                                                   outs[l], T);
    // f32 layer output -> bf16 A-operand for next GEMM (view (32,T/2,1024))
    cv(outs[l], XB, 32 * T * 512);
    T >>= 1;
    K = 1024;
  }
  const float* enc = outs[3];  // (32, 100, 1024) f32 ; XB = same data in bf16

  // ---- comp_listener = tanh(enc @ psi_W^T + psi_b) : (3200,128)
  k_psi_gemm<<<dim3(2, 200), dim3(128), 0, stream>>>(XB, psiB, psib, CL, 128, 1024);

  // ---- attention decoder (persistent, 100 steps)
  k_decoder<<<dim3(1), dim3(1024), SMEM_DEC, stream>>>(enc, CL, WdecB, dbih, dbhh, phiB,
                                                       charB, charb, embed, state_len,
                                                       decode_step, (float*)d_out);
}
